// MemoryGraph_38293928411456
// MI455X (gfx1250) — matrix operand order repaired and checked by reading, not compiled, not run on hardware
//
#include <hip/hip_runtime.h>
#include <hip/hip_bf16.h>
#include <cstdint>

#define B_    2
#define NC_   256
#define CN_   256
#define K_    16
#define DN_   64
#define G_    8
#define HM_   128
#define HS_   128
#define HMOD_ 64
#define MODO_ 5
#define F3_   192   // 3*DN

typedef __attribute__((ext_vector_type(16))) __bf16       v16bf;
typedef __attribute__((ext_vector_type(8)))  float        v8f;
typedef __attribute__((ext_vector_type(4)))  unsigned int u32x4;
typedef __attribute__((ext_vector_type(8)))  int          i32x8;
typedef __attribute__((ext_vector_type(4)))  int          i32x4;
typedef __attribute__((ext_vector_type(4)))  float        f32x4;
typedef __attribute__((ext_vector_type(2)))  float        f32x2;

#if defined(__has_builtin)
#if __has_builtin(__builtin_amdgcn_tensor_load_to_lds) && \
    __has_builtin(__builtin_amdgcn_s_wait_tensorcnt)
#define HAVE_TDM 1
#endif
#endif
#ifndef HAVE_TDM
#define HAVE_TDM 0
#endif

union FragBF { v16bf bf; u32x4 u[2]; };

__device__ __forceinline__ unsigned short f2bf(float f) {
  unsigned int u = __float_as_uint(f);
  u += 0x7FFFu + ((u >> 16) & 1u);           // round-to-nearest-even
  return (unsigned short)(u >> 16);
}
__device__ __forceinline__ unsigned int packbf(float a, float b) {
  return (unsigned int)f2bf(a) | ((unsigned int)f2bf(b) << 16);
}
// branch-free tanh: native V_TANH_F32 if the builtin exists, else exp2+rcp
__device__ __forceinline__ float tanh_fast(float z) {
#if defined(__has_builtin) && __has_builtin(__builtin_amdgcn_tanhf)
  return __builtin_amdgcn_tanhf(z);
#else
  float e = __builtin_amdgcn_exp2f(z * 2.885390081777927f);   // e^(2z)
  return 1.0f - 2.0f * __builtin_amdgcn_rcpf(e + 1.0f);
#endif
}
__device__ __forceinline__ float gelu_tanh(float x) {
  float x3 = x * x * x;
  return 0.5f * x * (1.0f + tanh_fast(0.7978845608028654f * (x + 0.044715f * x3)));
}

// A-fragment (16x32 bf16, M x K): lane<16 -> M=lane, K={k0..k0+7, k0+16..k0+23}
//                                 lane>=16 -> M=lane-16, K={k0+8..15, k0+24..31}
__device__ __forceinline__ v16bf fragA(const unsigned short* buf, int mbase, int ld,
                                       int k0, int lane) {
  int r  = mbase + (lane & 15);
  int kb = k0 + ((lane >> 4) << 3);
  const u32x4* p = (const u32x4*)(buf + r * ld + kb);
  FragBF f; f.u[0] = p[0]; f.u[1] = p[2];   // +0 and +16 halfwords
  return f.bf;
}
// B-fragment (32x16 bf16, K x N) from weights stored [N][K] (K contiguous):
// lane<16 -> N=lane, K={k0..k0+7, k0+16..k0+23} ; lane>=16 -> N=lane-16, K={k0+8..15, k0+24..31}
__device__ __forceinline__ v16bf fragB(const unsigned short* buf, int nbase, int ld,
                                       int k0, int lane) {
  int c  = nbase + (lane & 15);
  int kb = k0 + ((lane >> 4) << 3);
  const u32x4* p = (const u32x4*)(buf + c * ld + kb);
  FragBF f; f.u[0] = p[0]; f.u[1] = p[2];   // +0 and +16 halfwords
  return f.bf;
}
__device__ __forceinline__ v8f wmma_bf16(v16bf a, v16bf b, v8f c) {
  return __builtin_amdgcn_wmma_f32_16x16x32_bf16(false, a, false, b, (short)0, c,
                                                 false, false);
}

// ---- LDS layout (bytes) --------------------------------------------------
#define OFF_HF   0        // f32  h tile            256*64*4  = 65536
#define OFF_HB   65536    // bf16 h tile            256*64*2  = 32768
#define OFF_AB   98304    // bf16 agg -> msg        256*64*2  = 32768
#define OFF_NB   131072   // bf16 neuron_id -> ctx  256*64*2  = 32768
#define OFF_MHB  163840   // bf16 mh -> sh          256*128*2 = 65536
#define OFF_W1   229376   // bf16 w1 (msg->state)   128*192*2 = 49152
#define OFF_W2   278528   // bf16 w2 (msg->state)   64*128*2  = 16384
#define OFF_B1   294912   // f32 bias1 [128]
#define OFF_B2   295424   // f32 bias2 [64]
#define OFF_POOL 295680   // f32 pool  [128] (h sums | msg sums)
#define OFF_PH   296192   // f32 ph    [64]
#define SMEM_TOTAL 296448

__global__ void __launch_bounds__(256)
mem_graph_kernel(const float* __restrict__ h, const float* __restrict__ w_conn,
                 const float* __restrict__ cell_context, const float* __restrict__ neuron_id,
                 const float* __restrict__ msg_w1, const float* __restrict__ msg_b1,
                 const float* __restrict__ msg_w2, const float* __restrict__ msg_b2,
                 const float* __restrict__ state_w1, const float* __restrict__ state_b1,
                 const float* __restrict__ state_w2, const float* __restrict__ state_b2,
                 const float* __restrict__ mod_w1, const float* __restrict__ mod_b1,
                 const float* __restrict__ mod_w2, const float* __restrict__ mod_b2,
                 const int* __restrict__ conn_idx, const int* __restrict__ cell_to_group,
                 float* __restrict__ hOut, float* __restrict__ msgOut,
                 float* __restrict__ modOut) {
  extern __shared__ unsigned char smem[];
  float*          hF  = (float*)(smem + OFF_HF);
  unsigned short* hB  = (unsigned short*)(smem + OFF_HB);
  unsigned short* aB  = (unsigned short*)(smem + OFF_AB);
  unsigned short* nB  = (unsigned short*)(smem + OFF_NB);
  unsigned short* mhB = (unsigned short*)(smem + OFF_MHB);
  unsigned short* w1B = (unsigned short*)(smem + OFF_W1);
  unsigned short* w2B = (unsigned short*)(smem + OFF_W2);
  float*          b1L = (float*)(smem + OFF_B1);
  float*          b2L = (float*)(smem + OFF_B2);
  float*          pool = (float*)(smem + OFF_POOL);
  float*          phL  = (float*)(smem + OFF_PH);

  const int tid  = threadIdx.x;
  const int lane = tid & 31;
  const int wv   = tid >> 5;                 // 8 waves
  const int cell = blockIdx.x % NC_;
  const int bb   = blockIdx.x / NC_;
  const int g    = cell_to_group[cell];
  const size_t actBase = (((size_t)bb * NC_ + cell) * CN_) * DN_;

  // ---- Phase 0: stage h (f32 via TDM + bf16), neuron_id, msg weights -----
#if HAVE_TDM
  if (wv == 0) {
    // Tensor Data Mover: 1-row descriptor streaming the 64KB h tile -> hF.
    unsigned long long ga = (unsigned long long)(const void*)(h + actBase);
    unsigned lds_off = (unsigned)(unsigned long long)(void*)hF;   // flat[31:0] = LDS byte addr
    u32x4 g0;
    g0[0] = 1u;                                        // count=1 (valid user D#)
    g0[1] = lds_off;                                   // lds_addr
    g0[2] = (unsigned)(ga & 0xffffffffull);            // global_addr[31:0]
    g0[3] = (unsigned)((ga >> 32) & 0x01ffffffull) | (2u << 30);  // addr[56:32] | type=2
    i32x8 g1;
    g1[0] = (int)(2u << 16);          // data_size = 4B
    g1[1] = (int)(16384u << 16);      // tensor_dim0[15:0] in bits[63:48]
    g1[2] = (int)(1u << 16);          // tensor_dim0 hi = 0, tensor_dim1 = 1
    g1[3] = (int)(16384u << 16);      // tile_dim0 = 16384
    g1[4] = 0;                        // tile_dim1/2 unused
    g1[5] = 16384;                    // tensor_dim0_stride[31:0]
    g1[6] = 0;
    g1[7] = 0;
    i32x4 z4 = {0, 0, 0, 0};
#if __clang_major__ >= 23
    i32x8 z8 = {0, 0, 0, 0, 0, 0, 0, 0};
    __builtin_amdgcn_tensor_load_to_lds(g0, g1, z4, z4, z8, 0);
#else
    __builtin_amdgcn_tensor_load_to_lds(g0, g1, z4, z4, 0);
#endif
    __builtin_amdgcn_s_wait_tensorcnt(0);
  }
#endif
  {
    const f32x2* s2 = (const f32x2*)(h + actBase);
    unsigned int* hb2 = (unsigned int*)hB;
#if !HAVE_TDM
    f32x2* hf2 = (f32x2*)hF;
#endif
    for (int i = tid; i < CN_ * DN_ / 2; i += 256) {
      f32x2 v = s2[i];
#if !HAVE_TDM
      hf2[i] = v;
#endif
      hb2[i] = packbf(v.x, v.y);
    }
    const f32x2* nid2 = (const f32x2*)(neuron_id + (size_t)cell * CN_ * DN_);
    unsigned int* nb2 = (unsigned int*)nB;
    for (int i = tid; i < CN_ * DN_ / 2; i += 256) {
      f32x2 v = nid2[i];
      nb2[i] = packbf(v.x, v.y);
    }
    const f32x2* w1s = (const f32x2*)(msg_w1 + (size_t)g * HM_ * F3_);
    unsigned int* w1d = (unsigned int*)w1B;
    for (int i = tid; i < HM_ * F3_ / 2; i += 256) { f32x2 v = w1s[i]; w1d[i] = packbf(v.x, v.y); }
    const f32x2* w2s = (const f32x2*)(msg_w2 + (size_t)g * DN_ * HM_);
    unsigned int* w2d = (unsigned int*)w2B;
    for (int i = tid; i < DN_ * HM_ / 2; i += 256) { f32x2 v = w2s[i]; w2d[i] = packbf(v.x, v.y); }
    if (tid < HM_) b1L[tid] = msg_b1[g * HM_ + tid];
    if (tid < DN_) b2L[tid] = msg_b2[g * DN_ + tid];
    if (tid < 128) pool[tid] = 0.0f;
  }
  __syncthreads();

  // ---- Phase 0b: gather + weighted aggregation (1 thread = 1 neuron) -----
  {
    const int c = tid;
    const int*   ci = conn_idx + ((size_t)cell * CN_ + c) * K_;
    const float* wc = w_conn + (((size_t)bb * NC_ + cell) * CN_ + c) * K_;
    int   idx[K_];
    float wk[K_];
#pragma unroll
    for (int k2 = 0; k2 < K_; ++k2) { idx[k2] = ci[k2] * (DN_ / 4); wk[k2] = wc[k2]; }
    unsigned int* dst = (unsigned int*)(aB + c * DN_);
#pragma unroll
    for (int ch = 0; ch < 2; ++ch) {                    // two 32-dim chunks
      f32x4 acc[8] = {};
#pragma unroll
      for (int k2 = 0; k2 < K_; ++k2) {
        const f32x4* row = ((const f32x4*)hF) + idx[k2] + ch * 8;
        float wvv = wk[k2];
#pragma unroll
        for (int j = 0; j < 8; ++j) acc[j] += wvv * row[j];
      }
#pragma unroll
      for (int j = 0; j < 8; ++j) {
        f32x4 v = acc[j];
        dst[ch * 16 + 2 * j]     = packbf(v.x, v.y);
        dst[ch * 16 + 2 * j + 1] = packbf(v.z, v.w);
      }
    }
  }
  __syncthreads();

  // ---- Phase 1: GEMM1  mh = gelu([h|agg|nid] @ msg_w1^T + b1) ------------
  {
    v16bf bf[6];
#pragma unroll
    for (int ks = 0; ks < 6; ++ks) bf[ks] = fragB(w1B, wv * 16, F3_, ks * 32, lane);
    const int col = wv * 16 + (lane & 15);
    const float bias = b1L[col];
#pragma unroll 2
    for (int mt = 0; mt < 16; ++mt) {
      v16bf a0 = fragA(hB, mt * 16, DN_, 0,  lane);
      v16bf a1 = fragA(hB, mt * 16, DN_, 32, lane);
      v16bf a2 = fragA(aB, mt * 16, DN_, 0,  lane);
      v16bf a3 = fragA(aB, mt * 16, DN_, 32, lane);
      v16bf a4 = fragA(nB, mt * 16, DN_, 0,  lane);
      v16bf a5 = fragA(nB, mt * 16, DN_, 32, lane);
      v8f acc = {};
      acc = wmma_bf16(a0, bf[0], acc);
      acc = wmma_bf16(a1, bf[1], acc);
      acc = wmma_bf16(a2, bf[2], acc);
      acc = wmma_bf16(a3, bf[3], acc);
      acc = wmma_bf16(a4, bf[4], acc);
      acc = wmma_bf16(a5, bf[5], acc);
      int rbase = mt * 16 + ((lane >> 4) << 3);
#pragma unroll
      for (int r = 0; r < 8; ++r)
        mhB[(rbase + r) * HM_ + col] = f2bf(gelu_tanh(acc[r] + bias));
    }
  }
  __syncthreads();

  // ---- Phase 2: GEMM2  msg = mh @ msg_w2^T + b2 (emit f32 + bf16 + pool) -
  {
    v16bf bf[4];
    const int nt = wv & 3;
#pragma unroll
    for (int ks = 0; ks < 4; ++ks) bf[ks] = fragB(w2B, nt * 16, HM_, ks * 32, lane);
    const int col = nt * 16 + (lane & 15);
    const float bias = b2L[col];
    float colsum = 0.0f;
#pragma unroll 2
    for (int m = 0; m < 8; ++m) {
      int mt = ((wv >> 2) << 3) + m;
      v16bf a0 = fragA(mhB, mt * 16, HM_, 0,  lane);
      v16bf a1 = fragA(mhB, mt * 16, HM_, 32, lane);
      v16bf a2 = fragA(mhB, mt * 16, HM_, 64, lane);
      v16bf a3 = fragA(mhB, mt * 16, HM_, 96, lane);
      v8f acc = {};
      acc = wmma_bf16(a0, bf[0], acc);
      acc = wmma_bf16(a1, bf[1], acc);
      acc = wmma_bf16(a2, bf[2], acc);
      acc = wmma_bf16(a3, bf[3], acc);
      int rbase = mt * 16 + ((lane >> 4) << 3);
#pragma unroll
      for (int r = 0; r < 8; ++r) {
        float v = acc[r] + bias;
        msgOut[actBase + (size_t)(rbase + r) * DN_ + col] = v;
        aB[(rbase + r) * DN_ + col] = f2bf(v);
        colsum += v;
      }
    }
    atomicAdd(&pool[64 + col], colsum);
  }
  __syncthreads();

  // ---- Phase 2.5: stage state weights/biases, broadcast ctx into nB ------
  {
    const f32x2* w1s = (const f32x2*)(state_w1 + (size_t)g * HS_ * F3_);
    unsigned int* w1d = (unsigned int*)w1B;
    for (int i = tid; i < HS_ * F3_ / 2; i += 256) { f32x2 v = w1s[i]; w1d[i] = packbf(v.x, v.y); }
    const f32x2* w2s = (const f32x2*)(state_w2 + (size_t)g * DN_ * HS_);
    unsigned int* w2d = (unsigned int*)w2B;
    for (int i = tid; i < DN_ * HS_ / 2; i += 256) { f32x2 v = w2s[i]; w2d[i] = packbf(v.x, v.y); }
    if (tid < HS_) b1L[tid] = state_b1[g * HS_ + tid];
    if (tid < DN_) b2L[tid] = state_b2[g * DN_ + tid];
    const float* ctx = cell_context + ((size_t)bb * NC_ + cell) * DN_;
    unsigned int* nb2 = (unsigned int*)nB;
    for (int i = tid; i < CN_ * DN_ / 2; i += 256) {
      int d2 = i & 31;
      nb2[i] = packbf(ctx[2 * d2], ctx[2 * d2 + 1]);
    }
  }
  __syncthreads();

  // ---- Phase 3: GEMM3  sh = gelu([h|msg|ctx] @ state_w1^T + b1) ----------
  {
    v16bf bf[6];
#pragma unroll
    for (int ks = 0; ks < 6; ++ks) bf[ks] = fragB(w1B, wv * 16, F3_, ks * 32, lane);
    const int col = wv * 16 + (lane & 15);
    const float bias = b1L[col];
#pragma unroll 2
    for (int mt = 0; mt < 16; ++mt) {
      v16bf a0 = fragA(hB, mt * 16, DN_, 0,  lane);
      v16bf a1 = fragA(hB, mt * 16, DN_, 32, lane);
      v16bf a2 = fragA(aB, mt * 16, DN_, 0,  lane);
      v16bf a3 = fragA(aB, mt * 16, DN_, 32, lane);
      v16bf a4 = fragA(nB, mt * 16, DN_, 0,  lane);
      v16bf a5 = fragA(nB, mt * 16, DN_, 32, lane);
      v8f acc = {};
      acc = wmma_bf16(a0, bf[0], acc);
      acc = wmma_bf16(a1, bf[1], acc);
      acc = wmma_bf16(a2, bf[2], acc);
      acc = wmma_bf16(a3, bf[3], acc);
      acc = wmma_bf16(a4, bf[4], acc);
      acc = wmma_bf16(a5, bf[5], acc);
      int rbase = mt * 16 + ((lane >> 4) << 3);
#pragma unroll
      for (int r = 0; r < 8; ++r)
        mhB[(rbase + r) * HS_ + col] = f2bf(gelu_tanh(acc[r] + bias));
    }
  }
  __syncthreads();

  // ---- Phase 4: GEMM4  h_new = h + sh @ state_w2^T + b2 (emit + pool) ----
  {
    v16bf bf[4];
    const int nt = wv & 3;
#pragma unroll
    for (int ks = 0; ks < 4; ++ks) bf[ks] = fragB(w2B, nt * 16, HS_, ks * 32, lane);
    const int col = nt * 16 + (lane & 15);
    const float bias = b2L[col];
    float colsum = 0.0f;
#pragma unroll 2
    for (int m = 0; m < 8; ++m) {
      int mt = ((wv >> 2) << 3) + m;
      v16bf a0 = fragA(mhB, mt * 16, HS_, 0,  lane);
      v16bf a1 = fragA(mhB, mt * 16, HS_, 32, lane);
      v16bf a2 = fragA(mhB, mt * 16, HS_, 64, lane);
      v16bf a3 = fragA(mhB, mt * 16, HS_, 96, lane);
      v8f acc = {};
      acc = wmma_bf16(a0, bf[0], acc);
      acc = wmma_bf16(a1, bf[1], acc);
      acc = wmma_bf16(a2, bf[2], acc);
      acc = wmma_bf16(a3, bf[3], acc);
      int rbase = mt * 16 + ((lane >> 4) << 3);
#pragma unroll
      for (int r = 0; r < 8; ++r) {
        int rowi = rbase + r;
        float v = hF[rowi * DN_ + col] + acc[r] + bias;
        hOut[actBase + (size_t)rowi * DN_ + col] = v;
        colsum += v;
      }
    }
    atomicAdd(&pool[col], colsum);
  }
  __syncthreads();

  // ---- Phase 5: per-cell modulation MLP on pooled means ------------------
  if (tid < HMOD_) {
    const float* mw1 = mod_w1 + (size_t)cell * (2 * DN_) * HMOD_;
    float s = mod_b1[cell * HMOD_ + tid];
    const float inv = 1.0f / (float)CN_;
    for (int f = 0; f < 2 * DN_; ++f) s += (pool[f] * inv) * mw1[f * HMOD_ + tid];
    phL[tid] = gelu_tanh(s);
  }
  __syncthreads();
  if (tid < MODO_) {
    const float* mw2 = mod_w2 + (size_t)cell * HMOD_ * MODO_;
    float s = mod_b2[cell * MODO_ + tid];
    for (int hh = 0; hh < HMOD_; ++hh) s += phL[hh] * mw2[hh * MODO_ + tid];
    modOut[((size_t)bb * NC_ + cell) * MODO_ + tid] = s;
  }
}

extern "C" void kernel_launch(void* const* d_in, const int* in_sizes, int n_in,
                              void* d_out, int out_size, void* d_ws, size_t ws_size,
                              hipStream_t stream) {
  (void)in_sizes; (void)n_in; (void)out_size; (void)d_ws; (void)ws_size;
  const float* h            = (const float*)d_in[0];
  const float* w_conn       = (const float*)d_in[1];
  const float* cell_context = (const float*)d_in[2];
  const float* neuron_id    = (const float*)d_in[3];
  const float* msg_w1       = (const float*)d_in[4];
  const float* msg_b1       = (const float*)d_in[5];
  const float* msg_w2       = (const float*)d_in[6];
  const float* msg_b2       = (const float*)d_in[7];
  const float* state_w1     = (const float*)d_in[8];
  const float* state_b1     = (const float*)d_in[9];
  const float* state_w2     = (const float*)d_in[10];
  const float* state_b2     = (const float*)d_in[11];
  const float* mod_w1       = (const float*)d_in[12];
  const float* mod_b1       = (const float*)d_in[13];
  const float* mod_w2       = (const float*)d_in[14];
  const float* mod_b2       = (const float*)d_in[15];
  const int*   conn_idx     = (const int*)d_in[16];
  const int*   cell_to_grp  = (const int*)d_in[17];

  float* out    = (float*)d_out;
  float* hOut   = out;
  float* msgOut = out + (size_t)B_ * NC_ * CN_ * DN_;
  float* modOut = out + 2 * (size_t)B_ * NC_ * CN_ * DN_;

  hipFuncSetAttribute(reinterpret_cast<const void*>(&mem_graph_kernel),
                      hipFuncAttributeMaxDynamicSharedMemorySize, SMEM_TOTAL);

  mem_graph_kernel<<<dim3(B_ * NC_), dim3(256), SMEM_TOTAL, stream>>>(
      h, w_conn, cell_context, neuron_id,
      msg_w1, msg_b1, msg_w2, msg_b2,
      state_w1, state_b1, state_w2, state_b2,
      mod_w1, mod_b1, mod_w2, mod_b2,
      conn_idx, cell_to_grp, hOut, msgOut, modOut);
}
